// OriMultiHeadAttention_5428838662759
// MI455X (gfx1250) — compile-verified, matrix operand order checked by reading, not run on hardware
//
#include <hip/hip_runtime.h>
#include <hip/hip_bf16.h>

typedef __attribute__((ext_vector_type(16))) __bf16 v16bf;
typedef __attribute__((ext_vector_type(4)))  __bf16 v4bf;
typedef __attribute__((ext_vector_type(8)))  float  v8f;

// Problem constants (match reference)
#define BB   4
#define LL   2048
#define DD   1024
#define HH   8
#define DK   128
#define HD   1024   // H * DK

// ---------------------------------------------------------------------------
// Stage 1: f32 -> bf16 conversion (n is a multiple of 4)
// ---------------------------------------------------------------------------
__global__ __launch_bounds__(256) void cvt_f32_to_bf16(const float* __restrict__ in,
                                                       __bf16* __restrict__ out,
                                                       int n) {
  int i = (blockIdx.x * 256 + threadIdx.x) * 4;
  if (i >= n) return;
  float4 v = *reinterpret_cast<const float4*>(in + i);
  v4bf o;
  o[0] = (__bf16)v.x;
  o[1] = (__bf16)v.y;
  o[2] = (__bf16)v.z;
  o[3] = (__bf16)v.w;
  *reinterpret_cast<v4bf*>(out + i) = o;
}

// ---------------------------------------------------------------------------
// Stage 2: projection GEMM  Out[m,n] = sum_k X[m,k] * W[n,k] + bias[n]
// X: [8192, 1024] bf16 row-major, W: [1024, 1024] bf16 row-major (N x K)
// One wave -> one 16x16 C tile, K-loop of 32 via v_wmma_f32_16x16x32_bf16.
// Block = 8 waves covering a 32x64 tile. Grid = (8192/32, 1024/64).
// ---------------------------------------------------------------------------
__global__ __launch_bounds__(256) void proj_gemm_bf16(const __bf16* __restrict__ X,
                                                      const __bf16* __restrict__ W,
                                                      const float*  __restrict__ bias,
                                                      __bf16* __restrict__ Out) {
  const int K = DD;
  const int N = HD;
  const int wave = threadIdx.x >> 5;
  const int lane = threadIdx.x & 31;

  const int m0 = blockIdx.x * 32 + (wave >> 2) * 16;   // 2 waves along M
  const int n0 = blockIdx.y * 64 + (wave & 3) * 16;    // 4 waves along N

  const int l15   = lane & 15;
  const int klane = (lane >> 4) << 4;                  // 0 or 16

  const size_t arow = (size_t)(m0 + l15) * K;          // A: row m, K-contig
  const size_t brow = (size_t)(n0 + l15) * K;          // B: row n of W, K-contig

  v8f c = {};
  for (int k = 0; k < K; k += 32) {
    v16bf a = *reinterpret_cast<const v16bf*>(X + arow + k + klane);
    v16bf b = *reinterpret_cast<const v16bf*>(W + brow + k + klane);
    c = __builtin_amdgcn_wmma_f32_16x16x32_bf16(false, a, false, b,
                                                (short)0, c, false, false);
  }

  const int   ncol  = n0 + l15;
  const float bv    = bias[ncol];
  const int   mbase = m0 + ((lane >> 4) << 3);         // C layout: M = r + 8*(lane/16)
#pragma unroll
  for (int r = 0; r < 8; ++r) {
    float v = c[r] + bv;
    Out[(size_t)(mbase + r) * N + ncol] = (__bf16)v;
  }
}

// ---------------------------------------------------------------------------
// Stage 3: fused scores + mask + softmax, one block per (b, h, 16-row q strip)
// Qp/Kp: [B*L, HD] bf16 (head h occupies columns h*DK .. h*DK+127)
// out:   [B,H,L,L] f32
// ---------------------------------------------------------------------------
__global__ __launch_bounds__(256) void attn_fused(const __bf16* __restrict__ Qp,
                                                  const __bf16* __restrict__ Kp,
                                                  const int*    __restrict__ mask,
                                                  float* __restrict__ out) {
  __shared__ float s[16][LL];      // 128 KB score strip (<= 320KB/WGP)
  __shared__ float s_red[16][16];
  __shared__ float s_stat[16];

  const int h  = blockIdx.x;
  const int qt = blockIdx.y;
  const int b  = blockIdx.z;
  const int q0 = qt * 16;

  const int wave = threadIdx.x >> 5;
  const int lane = threadIdx.x & 31;
  const int l15   = lane & 15;
  const int klane = (lane >> 4) << 4;                  // 0 or 16
  const float scale = 0.088388347648318447f;           // 1/sqrt(128)

  const size_t qrow_base = ((size_t)b * LL + q0 + l15) * HD + (size_t)h * DK;

  // A fragments (q strip) are tile-invariant: load once.
  v16bf a0 = *reinterpret_cast<const v16bf*>(Qp + qrow_base + 0  + klane);
  v16bf a1 = *reinterpret_cast<const v16bf*>(Qp + qrow_base + 32 + klane);
  v16bf a2 = *reinterpret_cast<const v16bf*>(Qp + qrow_base + 64 + klane);
  v16bf a3 = *reinterpret_cast<const v16bf*>(Qp + qrow_base + 96 + klane);

  // 128 column tiles of 16; 8 waves round-robin
  for (int t = wave; t < LL / 16; t += 8) {
    const int c0 = t * 16;
    const size_t krow_base = ((size_t)b * LL + c0 + l15) * HD + (size_t)h * DK;

    v16bf b0 = *reinterpret_cast<const v16bf*>(Kp + krow_base + 0  + klane);
    v16bf b1 = *reinterpret_cast<const v16bf*>(Kp + krow_base + 32 + klane);
    v16bf b2 = *reinterpret_cast<const v16bf*>(Kp + krow_base + 64 + klane);
    v16bf b3 = *reinterpret_cast<const v16bf*>(Kp + krow_base + 96 + klane);

    v8f c = {};
    c = __builtin_amdgcn_wmma_f32_16x16x32_bf16(false, a0, false, b0, (short)0, c, false, false);
    c = __builtin_amdgcn_wmma_f32_16x16x32_bf16(false, a1, false, b1, (short)0, c, false, false);
    c = __builtin_amdgcn_wmma_f32_16x16x32_bf16(false, a2, false, b2, (short)0, c, false, false);
    c = __builtin_amdgcn_wmma_f32_16x16x32_bf16(false, a3, false, b3, (short)0, c, false, false);

    // Epilogue: scale + mask into LDS.  C layout: M = r + 8*(lane/16), N = lane&15
    const int ncol  = c0 + l15;
    const int mbase = (lane >> 4) << 3;
#pragma unroll
    for (int r = 0; r < 8; ++r) {
      const int qq = mbase + r;
      float v = c[r] * scale;
      const int mv = mask[((size_t)b * LL + q0 + qq) * LL + ncol];
      s[qq][ncol] = (mv == 0) ? -1.0e9f : v;
    }
  }
  __syncthreads();

  // Row softmax: 16 threads per row, float4 LDS traffic (ds_load_b128)
  const int row = threadIdx.x >> 4;
  const int sub = threadIdx.x & 15;
  float4* srow4 = reinterpret_cast<float4*>(&s[row][0]);   // 512 float4 per row

  float mx = -3.0e38f;
  for (int j = sub; j < LL / 4; j += 16) {
    float4 v = srow4[j];
    mx = fmaxf(mx, fmaxf(fmaxf(v.x, v.y), fmaxf(v.z, v.w)));
  }
  s_red[row][sub] = mx;
  __syncthreads();
  if (sub == 0) {
    float m2 = s_red[row][0];
#pragma unroll
    for (int i = 1; i < 16; ++i) m2 = fmaxf(m2, s_red[row][i]);
    s_stat[row] = m2;
  }
  __syncthreads();

  const float rowmax = s_stat[row];
  float sum = 0.0f;
  for (int j = sub; j < LL / 4; j += 16) {
    float4 v = srow4[j];
    v.x = __expf(v.x - rowmax);
    v.y = __expf(v.y - rowmax);
    v.z = __expf(v.z - rowmax);
    v.w = __expf(v.w - rowmax);
    srow4[j] = v;
    sum += v.x + v.y + v.z + v.w;
  }
  s_red[row][sub] = sum;
  __syncthreads();
  if (sub == 0) {
    float t2 = 0.0f;
#pragma unroll
    for (int i = 0; i < 16; ++i) t2 += s_red[row][i];
    s_stat[row] = 1.0f / t2;
  }
  __syncthreads();

  // Coalesced vectorized write of normalized probabilities (global_store_b128)
  const size_t obase = (((size_t)b * HH + h) * LL + q0) * (size_t)LL;
  const float4* sall4 = reinterpret_cast<const float4*>(&s[0][0]);
  for (int idx = threadIdx.x; idx < 16 * LL / 4; idx += 256) {
    const int rr = idx >> 9;             // / (2048/4)
    float4 v = sall4[idx];
    const float inv = s_stat[rr];
    v.x *= inv; v.y *= inv; v.z *= inv; v.w *= inv;
    // s is [16][LL] contiguous; out rows are LL apart with the same layout
    *reinterpret_cast<float4*>(out + obase + (size_t)rr * LL + ((idx & 511) << 2)) = v;
  }
}

// ---------------------------------------------------------------------------
// Host launcher
// ---------------------------------------------------------------------------
extern "C" void kernel_launch(void* const* d_in, const int* in_sizes, int n_in,
                              void* d_out, int out_size, void* d_ws, size_t ws_size,
                              hipStream_t stream) {
  const float* query   = (const float*)d_in[0];
  const float* key_inp = (const float*)d_in[1];
  const int*   mask    = (const int*)d_in[2];
  const float* Wr0     = (const float*)d_in[3];
  const float* br0     = (const float*)d_in[4];
  const float* Wi1     = (const float*)d_in[9];
  const float* bi1     = (const float*)d_in[10];
  float* out = (float*)d_out;

  char* ws = (char*)d_ws;
  const size_t nQ = (size_t)BB * LL * DD;   // 8388608
  const size_t nW = (size_t)HD * DD;        // 1048576

  __bf16* qbf   = (__bf16*)(ws);
  __bf16* kbf   = (__bf16*)(ws + nQ * 2);
  __bf16* wr0bf = (__bf16*)(ws + nQ * 4);
  __bf16* wi1bf = (__bf16*)(ws + nQ * 4 + nW * 2);
  __bf16* qproj = (__bf16*)(ws + nQ * 4 + nW * 4);
  __bf16* kproj = (__bf16*)(ws + nQ * 6 + nW * 4);
  // total workspace: nQ*8 + nW*4 bytes = 68 MB

  // Stage 1: conversions
  cvt_f32_to_bf16<<<(int)(nQ / 4 / 256), 256, 0, stream>>>(query,   qbf,   (int)nQ);
  cvt_f32_to_bf16<<<(int)(nQ / 4 / 256), 256, 0, stream>>>(key_inp, kbf,   (int)nQ);
  cvt_f32_to_bf16<<<(int)(nW / 4 / 256), 256, 0, stream>>>(Wr0,     wr0bf, (int)nW);
  cvt_f32_to_bf16<<<(int)(nW / 4 / 256), 256, 0, stream>>>(Wi1,     wi1bf, (int)nW);

  // Stage 2: projections (M=8192, N=1024, K=1024)
  dim3 gp(BB * LL / 32, HD / 64);
  proj_gemm_bf16<<<gp, 256, 0, stream>>>(qbf, wr0bf, br0, qproj);
  proj_gemm_bf16<<<gp, 256, 0, stream>>>(kbf, wi1bf, bi1, kproj);

  // Stage 3: fused attention. h fastest in grid -> mask rows reused in L2 across heads.
  dim3 ga(HH, LL / 16, BB);
  attn_fused<<<ga, 256, 0, stream>>>(qproj, kproj, mask, out);
}
